// Attention_57406532878693
// MI455X (gfx1250) — matrix operand order repaired and checked by reading, not compiled, not run on hardware
//
#include <hip/hip_runtime.h>

typedef __attribute__((ext_vector_type(16))) _Float16 v16h;
typedef __attribute__((ext_vector_type(8)))  float    v8f;
typedef __attribute__((ext_vector_type(4)))  float    v4f_;
typedef __attribute__((ext_vector_type(4)))  unsigned int v4u;
typedef __attribute__((ext_vector_type(8)))  int      v8i_;
typedef __attribute__((ext_vector_type(4)))  int      v4i_;

#define T_DIM 2048
#define B_DIM 32
#define D_DIM 1024
#define H_DIM 8
#define N_DIM 128
#define K2_DIM 16

// dynamic LDS layout (float offsets)
#define XSTAGE_OFF 0              // 8 waves * 2 bufs * (16*128) = 32768
#define GATES_OFF  32768          // 2048 * 16 = 32768
#define ATT_OFF    65536          // 2048
#define MSUM_OFF   67584          // 128
#define HV_OFF     67712          // 16
#define MISC_OFF   67728          // 8
#define RED_OFF    67736          // 256
#define SMEM_FLOATS 67992         // ~266 KB of the 320 KB WGP LDS

#define HAVE_TDM (__has_builtin(__builtin_amdgcn_tensor_load_to_lds) && \
                  __has_builtin(__builtin_amdgcn_s_wait_tensorcnt))

__global__ void __launch_bounds__(256)
attn_pool_kernel(const float* __restrict__ hyp,
                 const float* __restrict__ Ww,  const float* __restrict__ Wb,
                 const float* __restrict__ Wmw, const float* __restrict__ Wmb,
                 const float* __restrict__ Whw, const float* __restrict__ Whb,
                 float* __restrict__ out)
{
    extern __shared__ float smem[];
    float* xstage = smem + XSTAGE_OFF;
    float* gates  = smem + GATES_OFF;
    float* att    = smem + ATT_OFF;
    float* msum   = smem + MSUM_OFF;
    float* hv     = smem + HV_OFF;
    float* misc   = smem + MISC_OFF;
    float* red    = smem + RED_OFF;

    const int tid  = threadIdx.x;
    const int lane = tid & 31;          // wave32
    const int wv   = tid >> 5;          // 8 waves
    const int bh   = blockIdx.x;        // 256 blocks = B*H
    const int b    = bh >> 3;
    const int h    = bh & 7;

    if (tid < N_DIM) msum[tid] = 0.0f;
    __syncthreads();

    // ---- B fragments: W_w^T (128x16) as f16, resident in VGPRs --------------
    // B 32x16 f16, contiguous k order used here (same as A): lane%16 = N column; lanes 0-15 hold K=0..15 of the
    // chunk, lanes 16-31 hold K=16..31 (2 halves per VGPR).
    const int nsel = lane & 15;
    const int hi   = lane >> 4;
    v16h bfrag[4];
    #pragma unroll
    for (int c = 0; c < 4; ++c) {
        #pragma unroll
        for (int i = 0; i < 16; ++i)
            bfrag[c][i] = (_Float16)Ww[nsel * N_DIM + 32 * c + hi * 16 + i];
    }
    const float wbias = Wb[nsel];

    const float* xbase = hyp + (size_t)b * D_DIM + (size_t)h * N_DIM; // + t*B*D
    float* xbuf0 = xstage + wv * 4096;          // two 16x128 f32 tiles per wave
    float* xbuf1 = xbuf0 + 2048;

    float meanacc[4] = {0.f, 0.f, 0.f, 0.f};

#if HAVE_TDM
    // ---- TDM issue: one tensor_load_to_lds per 16x128 tile ------------------
    // D# group0: count=1, lds_addr, 57b global_addr, type=2 ("image").
    // D# group1: data_size=4B, tensor_dim0=128, tensor_dim1=16,
    //            tile_dim0=128, tile_dim1=16, tensor_dim0_stride=B*D.
    auto issue_tile = [&](int jt, float* dstbuf) {
        const size_t gaddr = (size_t)(const void*)(xbase + (size_t)jt * 16 * (B_DIM * D_DIM));
        const unsigned lds_off = (unsigned)(size_t)(void*)dstbuf; // LDS aperture low bits
        v4u g0;
        g0[0] = 1u;                                   // count=1, user descriptor
        g0[1] = lds_off;                              // lds_addr (bytes)
        g0[2] = (unsigned)(gaddr & 0xFFFFFFFFu);      // global_addr[31:0]
        g0[3] = (unsigned)((gaddr >> 32) & 0x01FFFFFFu) | 0x80000000u; // [56:32] | type=2
        v8i_ g1;
        g1[0] = (int)(2u << 16);                      // data_size = 4 bytes
        g1[1] = (int)(128u << 16);                    // tensor_dim0 = 128
        g1[2] = (int)(16u << 16);                     // tensor_dim1 = 16
        g1[3] = (int)(128u << 16);                    // tile_dim0 = 128
        g1[4] = 16;                                   // tile_dim1 = 16, tile_dim2 = 0
        g1[5] = (int)(B_DIM * D_DIM);                 // tensor_dim0_stride lo32
        g1[6] = 0;                                    // stride0 hi, stride1 lo
        g1[7] = 0;
        const v4i_ gz4 = {0, 0, 0, 0};                // groups 2/3 unused (2D tile)
        const v8i_ gz8 = {0, 0, 0, 0, 0, 0, 0, 0};
        __builtin_amdgcn_tensor_load_to_lds(g0, g1, gz4, gz4, gz8, 0);
    };
    issue_tile(wv, xbuf0);                            // prologue: first tile
#endif

    // ---- Stage A: gate GEMM via WMMA + mean partials, TDM double-buffered ----
    for (int i = 0; i < T_DIM / 16 / 8; ++i) {        // 16 tiles per wave
        const int jt = wv + 8 * i;
        const int t0 = jt * 16;
        float* cur = (i & 1) ? xbuf1 : xbuf0;
#if HAVE_TDM
        if (i < 15) {
            issue_tile(jt + 8, (i & 1) ? xbuf0 : xbuf1);
            __builtin_amdgcn_s_wait_tensorcnt(1);     // current tile landed
        } else {
            __builtin_amdgcn_s_wait_tensorcnt(0);
        }
#else
        // fallback: coalesced manual staging of the 16x128 tile
        #pragma unroll
        for (int r = 0; r < 16; ++r) {
            const v4f_ v = *(const v4f_*)(xbase +
                    (size_t)(t0 + r) * (B_DIM * D_DIM) + lane * 4);
            *(v4f_*)(cur + r * 128 + lane * 4) = v;
        }
#endif
        v8f acc = {};
        #pragma unroll
        for (int c = 0; c < 4; ++c) {
            // per-column partial sums for the time-mean
            float csum = 0.f;
            #pragma unroll
            for (int r = 0; r < 16; ++r) csum += cur[r * 128 + 32 * c + lane];
            meanacc[c] += csum;
            // A fragment: 16x32 A in contiguous k order, same as bfrag (lane%16 = M, lane/16 = K half)
            const float* ab = cur + nsel * 128 + 32 * c + hi * 16;
            const v4f_ q0 = *(const v4f_*)(ab);
            const v4f_ q1 = *(const v4f_*)(ab + 4);
            const v4f_ q2 = *(const v4f_*)(ab + 8);
            const v4f_ q3 = *(const v4f_*)(ab + 12);
            v16h a;
            a[0]=(_Float16)q0[0];  a[1]=(_Float16)q0[1];  a[2]=(_Float16)q0[2];  a[3]=(_Float16)q0[3];
            a[4]=(_Float16)q1[0];  a[5]=(_Float16)q1[1];  a[6]=(_Float16)q1[2];  a[7]=(_Float16)q1[3];
            a[8]=(_Float16)q2[0];  a[9]=(_Float16)q2[1];  a[10]=(_Float16)q2[2]; a[11]=(_Float16)q2[3];
            a[12]=(_Float16)q3[0]; a[13]=(_Float16)q3[1]; a[14]=(_Float16)q3[2]; a[15]=(_Float16)q3[3];
            acc = __builtin_amdgcn_wmma_f32_16x16x32_f16(
                    false, a, false, bfrag[c], (short)0, acc, false, false);
        }
        // D layout: VGPR r -> M = r + 8*hi, N = lane%16 ; t = t0 + M
        #pragma unroll
        for (int r = 0; r < 8; ++r)
            gates[(t0 + hi * 8 + r) * 16 + nsel] = tanhf(acc[r] + wbias);
    }
    #pragma unroll
    for (int c = 0; c < 4; ++c)
        atomicAdd(&msum[32 * c + lane], meanacc[c]);   // ds_add_f32
    __syncthreads();

    // ---- Stage B: gate_m, logits, softmax over T -----------------------------
    if (tid < K2_DIM) {
        float d = Wmb[tid];
        const float invT = 1.0f / (float)T_DIM;
        for (int k = 0; k < N_DIM; ++k)
            d += Wmw[tid * N_DIM + k] * (msum[k] * invT);
        hv[tid] = tanhf(d) * Whw[tid];   // fold gate_m * Wh_w
    }
    __syncthreads();

    const float whb = Whb[0];
    float lg[8];
    float lmax = -3.4e38f;
    #pragma unroll
    for (int j = 0; j < 8; ++j) {
        const int t = tid + 256 * j;
        float s = whb;
        #pragma unroll
        for (int n = 0; n < 16; ++n) s += gates[t * 16 + n] * hv[n];
        lg[j] = s;
        lmax = fmaxf(lmax, s);
    }
    red[tid] = lmax; __syncthreads();
    for (int s = 128; s > 0; s >>= 1) {
        if (tid < s) red[tid] = fmaxf(red[tid], red[tid + s]);
        __syncthreads();
    }
    const float gmax = red[0];
    __syncthreads();
    float lsum = 0.f;
    #pragma unroll
    for (int j = 0; j < 8; ++j) {
        const int t = tid + 256 * j;
        const float e = __expf(lg[j] - gmax);
        att[t] = e;
        lsum += e;
    }
    red[tid] = lsum; __syncthreads();
    for (int s = 128; s > 0; s >>= 1) {
        if (tid < s) red[tid] += red[tid + s];
        __syncthreads();
    }
    if (tid == 0) misc[0] = 1.0f / red[0];
    __syncthreads();

    // ---- Stage C: c = sum_t a_t * x_t  (second NT stream of hyp, b128) -------
    const int colg = (tid & 31) * 4;     // 4 columns per lane
    const int tg   = tid >> 5;           // 8 timestep groups
    const float* xq = xbase + colg;
    v4f_ acc4 = {0.f, 0.f, 0.f, 0.f};
    for (int j = 0; j < T_DIM / 8; ++j) {
        const int t = 8 * j + tg;
        if (t + 64 < T_DIM)
            __builtin_prefetch(xq + (size_t)(t + 64) * (B_DIM * D_DIM), 0, 1);
        const v4f_ v = __builtin_nontemporal_load(
                (const v4f_*)(xq + (size_t)t * (B_DIM * D_DIM)));
        acc4 += att[t] * v;
    }
    float* scr = xstage;                 // reuse staging LDS for reduction
    ((v4f_*)scr)[tid] = acc4;
    __syncthreads();
    if (tid < 128) {
        const int q = tid >> 2, r = tid & 3;
        float s = 0.f;
        #pragma unroll
        for (int g = 0; g < 8; ++g) s += scr[(g * 32 + q) * 4 + r];
        out[b * D_DIM + h * N_DIM + tid] = s * misc[0];
    }
}

extern "C" void kernel_launch(void* const* d_in, const int* in_sizes, int n_in,
                              void* d_out, int out_size, void* d_ws, size_t ws_size,
                              hipStream_t stream) {
    const float* hyp = (const float*)d_in[0];
    const float* Ww  = (const float*)d_in[1];
    const float* Wb  = (const float*)d_in[2];
    const float* Wmw = (const float*)d_in[3];
    const float* Wmb = (const float*)d_in[4];
    const float* Whw = (const float*)d_in[5];
    const float* Whb = (const float*)d_in[6];
    float* out = (float*)d_out;
    (void)in_sizes; (void)n_in; (void)out_size; (void)d_ws; (void)ws_size;

    dim3 grid(B_DIM * H_DIM);
    dim3 block(256);
    size_t shmem = (size_t)SMEM_FLOATS * sizeof(float);
    attn_pool_kernel<<<grid, block, shmem, stream>>>(hyp, Ww, Wb, Wmw, Wmb, Whw, Whb, out);
}
